// RecursiveMamba2_v34_43250320670943
// MI455X (gfx1250) — matrix operand order repaired and checked by reading, not compiled, not run on hardware
//
#include <hip/hip_runtime.h>
#include <hip/hip_bf16.h>

// ---------------------------------------------------------------------------
// RecursiveMamba2 for gfx1250 (MI455X): WMMA f16 GEMMs (register-blocked
// 32x64 wave tiles) + VALU selective scan.
// ---------------------------------------------------------------------------

#define D_MODEL   768
#define D_STATE   64
#define D_CONV    4
#define D_INNER   1536
#define HEADDIM   64
#define NHEADS    24          // D_INNER / HEADDIM
#define RANK      8
#define LORA_SCALE 2.0f       // 16.0 / RANK
#define D_IN_PROJ 3224        // 2*D_INNER + 2*D_STATE + NHEADS
#define D_IN_PAD  3264        // padded to multiple of 64 for WMMA macro-tile N
#define CONV_DIM  1664        // D_INNER + 2*D_STATE
#define BATCH     2
#define SEQLEN    1024
#define NTOK      (BATCH * SEQLEN)   // 2048
#define EPSF      1e-6f
#define MAX_LOOPS 8

typedef _Float16 f16_t;
typedef __attribute__((ext_vector_type(16))) _Float16 v16h;
typedef __attribute__((ext_vector_type(8)))  _Float16 v8h;
typedef __attribute__((ext_vector_type(8)))  float    v8f;

// ---------------------------------------------------------------------------
// helpers
// ---------------------------------------------------------------------------
__device__ __forceinline__ float silu_f(float v) {
  return v / (1.0f + __expf(-v));
}

__device__ __forceinline__ float block_reduce_sum(float v, float* smem) {
  // wave32 reduction then cross-wave via LDS
  #pragma unroll
  for (int o = 16; o > 0; o >>= 1) v += __shfl_xor(v, o, 32);
  const int lane = threadIdx.x & 31;
  const int wave = threadIdx.x >> 5;
  if (lane == 0) smem[wave] = v;
  __syncthreads();
  const int nw = (blockDim.x + 31) >> 5;
  float r = (threadIdx.x < (unsigned)nw) ? smem[threadIdx.x] : 0.0f;
  if (wave == 0) {
    #pragma unroll
    for (int o = 16; o > 0; o >>= 1) r += __shfl_xor(r, o, 32);
    if (lane == 0) smem[0] = r;
  }
  __syncthreads();
  return smem[0];
}

// ---------------------------------------------------------------------------
// weight prep: W = base + SCALE * (loraB @ loraA), cast to f16 (run once)
// ---------------------------------------------------------------------------
__global__ void prep_win_kernel(const float* __restrict__ base,
                                const float* __restrict__ lA,   // [RANK, 768]
                                const float* __restrict__ lB,   // [D_IN_PROJ, RANK]
                                f16_t* __restrict__ W) {        // [D_IN_PAD, 768]
  int idx = blockIdx.x * 256 + threadIdx.x;
  if (idx >= D_IN_PAD * D_MODEL) return;
  int r = idx / D_MODEL, c = idx - r * D_MODEL;
  float v = 0.0f;
  if (r < D_IN_PROJ) {
    v = base[(size_t)r * D_MODEL + c];
    #pragma unroll
    for (int j = 0; j < RANK; j++)
      v += LORA_SCALE * lB[r * RANK + j] * lA[j * D_MODEL + c];
  }
  W[idx] = (f16_t)v;
}

__global__ void prep_wout_kernel(const float* __restrict__ base,
                                 const float* __restrict__ lA,  // [RANK, 1536]
                                 const float* __restrict__ lB,  // [768, RANK]
                                 f16_t* __restrict__ W) {       // [768, 1536]
  int idx = blockIdx.x * 256 + threadIdx.x;
  if (idx >= D_MODEL * D_INNER) return;
  int r = idx / D_INNER, c = idx - r * D_INNER;
  float v = base[(size_t)r * D_INNER + c];
  #pragma unroll
  for (int j = 0; j < RANK; j++)
    v += LORA_SCALE * lB[r * RANK + j] * lA[j * D_INNER + c];
  W[idx] = (f16_t)v;
}

// ---------------------------------------------------------------------------
// rmsnorm + rope, fp32 -> f16 activations for GEMM1
// grid = NTOK blocks, 384 threads (one thread per dim-pair)
// ---------------------------------------------------------------------------
__global__ void rmsnorm_rope_kernel(const float* __restrict__ xin,
                                    const float* __restrict__ w,
                                    f16_t* __restrict__ hout,
                                    int loop, const int* __restrict__ n_loops) {
  if (loop >= n_loops[0]) return;
  const int tok = blockIdx.x;
  const int t   = threadIdx.x;          // 0..383 -> pair index
  const float* row = xin + (size_t)tok * D_MODEL;
  float x0 = row[2 * t], x1 = row[2 * t + 1];
  __shared__ float sred[16];
  float ss = block_reduce_sum(x0 * x0 + x1 * x1, sred);
  float rstd = rsqrtf(ss * (1.0f / D_MODEL) + EPSF);
  float n0 = x0 * rstd * w[2 * t];
  float n1 = x1 * rstd * w[2 * t + 1];
  // inv_freq = 10000^(-2t/768) = exp2(-(2t/768)*log2(10000))
  float inv_freq = exp2f(-(float)t * (2.0f / (float)D_MODEL) * 13.2877123795f);
  float ang = (float)loop * inv_freq;
  float s = __sinf(ang);
  float c = __cosf(ang);
  f16_t* orow = hout + (size_t)tok * D_MODEL;
  orow[2 * t]     = (f16_t)(n0 * c - n1 * s);
  orow[2 * t + 1] = (f16_t)(n1 * c + n0 * s);
}

// ---------------------------------------------------------------------------
// WMMA f16 GEMM:  C[M,N] = A[M,K] @ Bmat[N,K]^T   (both operands K-contiguous)
// One wave per block; wave computes a 32(M) x 64(N) macro-tile:
// 2 A-frags x 4 B-frags -> 8 independent 16x16 accumulators.
// grid = (N/64, M/32), block = 32 threads.
// ---------------------------------------------------------------------------
__global__ __launch_bounds__(32)
void gemm_wmma_f16_kernel(const f16_t* __restrict__ A,
                          const f16_t* __restrict__ Bmat,
                          float* __restrict__ C,
                          int N_, int K,
                          int loop, const int* __restrict__ n_loops) {
  if (loop >= n_loops[0]) return;
  const int lane = threadIdx.x & 31;
  const int m0   = blockIdx.y * 32;
  const int n0   = blockIdx.x * 64;
  const int lm   = lane & 15;
  const int hi   = lane >> 4;           // 0 for lanes 0-15, 1 for lanes 16-31

  // A fragment rows: row m; per-lane K chunks {hi*16..+7} and {hi*16+8..+15}
  const f16_t* arow[2];
  #pragma unroll
  for (int mi = 0; mi < 2; mi++)
    arow[mi] = A + (size_t)(m0 + mi * 16 + lm) * K + hi * 16;
  // B fragment columns: col n; per-lane 16 contiguous K at hi*16
  const f16_t* brow[4];
  #pragma unroll
  for (int ni = 0; ni < 4; ni++)
    brow[ni] = Bmat + (size_t)(n0 + ni * 16 + lm) * K + hi * 16;

  v8f acc[8];
  #pragma unroll
  for (int t = 0; t < 8; t++) acc[t] = (v8f){};

  for (int k = 0; k < K; k += 32) {
    v16h afrag[2], bfrag[4];
    #pragma unroll
    for (int mi = 0; mi < 2; mi++) {
      v8h lo  = *(const v8h*)(arow[mi] + k);
      v8h hi8 = *(const v8h*)(arow[mi] + k + 8);
      afrag[mi] = __builtin_shufflevector(lo, hi8,
                                          0, 1, 2, 3, 4, 5, 6, 7,
                                          8, 9, 10, 11, 12, 13, 14, 15);
    }
    #pragma unroll
    for (int ni = 0; ni < 4; ni++)
      bfrag[ni] = *(const v16h*)(brow[ni] + k);
    #pragma unroll
    for (int mi = 0; mi < 2; mi++)
      #pragma unroll
      for (int ni = 0; ni < 4; ni++)
        acc[mi * 4 + ni] = __builtin_amdgcn_wmma_f32_16x16x32_f16(
            /*neg_a=*/false, afrag[mi], /*neg_b=*/false, bfrag[ni],
            /*c_mod=*/(short)0, acc[mi * 4 + ni],
            /*reuse_a=*/false, /*reuse_b=*/false);
  }

  // D layout: vgpr v -> m = v + hi*8, n = lm
  #pragma unroll
  for (int mi = 0; mi < 2; mi++) {
    #pragma unroll
    for (int ni = 0; ni < 4; ni++) {
      float* crow = C + (size_t)(m0 + mi * 16 + hi * 8) * N_ + (n0 + ni * 16 + lm);
      v8f a = acc[mi * 4 + ni];
      #pragma unroll
      for (int v = 0; v < 8; v++) crow[(size_t)v * N_] = a[v];
    }
  }
}

// ---------------------------------------------------------------------------
// dt = softplus(zx[..., 3200+h] + dt_bias);  dA = exp(dt * -exp(A_log))
// ---------------------------------------------------------------------------
__global__ void dt_da_kernel(const float* __restrict__ zx,
                             const float* __restrict__ dt_bias,
                             const float* __restrict__ A_log,
                             float* __restrict__ dtb, float* __restrict__ dAb,
                             int loop, const int* __restrict__ n_loops) {
  if (loop >= n_loops[0]) return;
  int idx = blockIdx.x * 256 + threadIdx.x;
  if (idx >= NTOK * NHEADS) return;
  int h = idx % NHEADS, tok = idx / NHEADS;
  float v = zx[(size_t)tok * D_IN_PAD + (D_INNER + CONV_DIM) + h] + dt_bias[h];
  float dt = (v > 20.0f) ? v : log1pf(__expf(v));
  float A = -__expf(A_log[h]);
  dtb[idx] = dt;
  dAb[idx] = __expf(dt * A);
}

// ---------------------------------------------------------------------------
// depthwise causal conv(4) + silu over the xBC slice of zxbcdt
// ---------------------------------------------------------------------------
__global__ void conv_silu_kernel(const float* __restrict__ zx,
                                 const float* __restrict__ conv_w,
                                 const float* __restrict__ conv_b,
                                 float* __restrict__ xBC,
                                 int loop, const int* __restrict__ n_loops) {
  if (loop >= n_loops[0]) return;
  int idx = blockIdx.x * 256 + threadIdx.x;
  if (idx >= NTOK * CONV_DIM) return;
  int c   = idx % CONV_DIM;
  int tok = idx / CONV_DIM;
  int l = tok & (SEQLEN - 1);
  int b = tok >> 10;
  float acc = conv_b[c];
  #pragma unroll
  for (int k = 0; k < D_CONV; k++) {
    int ls = l - (D_CONV - 1) + k;
    if (ls >= 0)
      acc += zx[((size_t)(b * SEQLEN + ls)) * D_IN_PAD + D_INNER + c] *
             conv_w[c * D_CONV + k];
  }
  xBC[idx] = silu_f(acc);
}

// ---------------------------------------------------------------------------
// sequential selective scan. grid = BATCH*NHEADS blocks, 64 threads.
// thread p owns state row h[p][0..63] in registers; B/C broadcast via LDS.
// ---------------------------------------------------------------------------
__global__ __launch_bounds__(64)
void ssm_scan_kernel(const float* __restrict__ xBC,
                     const float* __restrict__ dtb,
                     const float* __restrict__ dAb,
                     const float* __restrict__ D_param,
                     float* __restrict__ ybuf,
                     int loop, const int* __restrict__ n_loops) {
  if (loop >= n_loops[0]) return;
  const int b    = blockIdx.x / NHEADS;
  const int head = blockIdx.x % NHEADS;
  const int p    = threadIdx.x;          // 0..63
  __shared__ float sB[D_STATE], sC[D_STATE];
  float hreg[D_STATE];
  #pragma unroll
  for (int n = 0; n < D_STATE; n++) hreg[n] = 0.0f;
  const float Dp = D_param[head];
  for (int l = 0; l < SEQLEN; l++) {
    const size_t tok = (size_t)b * SEQLEN + l;
    const float* row = xBC + tok * CONV_DIM;
    sB[p] = row[D_INNER + p];
    sC[p] = row[D_INNER + D_STATE + p];
    __syncthreads();
    const float xp  = row[head * HEADDIM + p];
    const float dtl = dtb[tok * NHEADS + head];
    const float dAl = dAb[tok * NHEADS + head];
    const float coef = dtl * xp;
    float y = 0.0f;
    #pragma unroll
    for (int n = 0; n < D_STATE; n++) {
      hreg[n] = fmaf(coef, sB[n], dAl * hreg[n]);
      y = fmaf(sC[n], hreg[n], y);
    }
    ybuf[tok * D_INNER + head * HEADDIM + p] = fmaf(Dp, xp, y);
    __syncthreads();
  }
}

// ---------------------------------------------------------------------------
// y = rmsnorm(y * silu(z), gated_norm_w) -> f16 for GEMM2
// grid = NTOK blocks, 256 threads, 6 elements each (1536 dims)
// ---------------------------------------------------------------------------
__global__ void gate_norm_kernel(const float* __restrict__ ybuf,
                                 const float* __restrict__ zx,
                                 const float* __restrict__ gw,
                                 f16_t* __restrict__ yg,
                                 int loop, const int* __restrict__ n_loops) {
  if (loop >= n_loops[0]) return;
  const int tok = blockIdx.x;
  const float* yrow = ybuf + (size_t)tok * D_INNER;
  const float* zrow = zx + (size_t)tok * D_IN_PAD;   // z = first 1536 cols
  float vals[6];
  float ss = 0.0f;
  #pragma unroll
  for (int j = 0; j < 6; j++) {
    int i = threadIdx.x * 6 + j;
    float v = yrow[i] * silu_f(zrow[i]);
    vals[j] = v;
    ss += v * v;
  }
  __shared__ float sred[8];
  float tot = block_reduce_sum(ss, sred);
  float rstd = rsqrtf(tot * (1.0f / D_INNER) + EPSF);
  f16_t* orow = yg + (size_t)tok * D_INNER;
  #pragma unroll
  for (int j = 0; j < 6; j++) {
    int i = threadIdx.x * 6 + j;
    orow[i] = (f16_t)(vals[j] * rstd * gw[i]);
  }
}

// ---------------------------------------------------------------------------
// x_cur = gemm2_out + lifeline_gate * x_prompt
// ---------------------------------------------------------------------------
__global__ void residual_kernel(const float* __restrict__ g2,
                                const float* __restrict__ xp,
                                const float* __restrict__ gate,
                                float* __restrict__ xcur,
                                int loop, const int* __restrict__ n_loops) {
  if (loop >= n_loops[0]) return;
  int idx = blockIdx.x * 256 + threadIdx.x;
  if (idx >= NTOK * D_MODEL) return;
  int c = idx % D_MODEL;
  xcur[idx] = g2[idx] + gate[c] * xp[idx];
}

__global__ void finalize_kernel(const float* __restrict__ xcur,
                                const float* __restrict__ xin,
                                float* __restrict__ out,
                                const int* __restrict__ n_loops) {
  int idx = blockIdx.x * 256 + threadIdx.x;
  if (idx >= NTOK * D_MODEL) return;
  out[idx] = (n_loops[0] > 0) ? xcur[idx] : xin[idx];
}

// ---------------------------------------------------------------------------
// host launcher
// ---------------------------------------------------------------------------
extern "C" void kernel_launch(void* const* d_in, const int* in_sizes, int n_in,
                              void* d_out, int out_size, void* d_ws, size_t ws_size,
                              hipStream_t stream) {
  const float* x          = (const float*)d_in[0];
  const float* x_prompt   = (const float*)d_in[1];
  const float* in_base    = (const float*)d_in[2];
  const float* lora_A_in  = (const float*)d_in[3];
  const float* lora_B_in  = (const float*)d_in[4];
  const float* conv_w     = (const float*)d_in[5];
  const float* conv_b     = (const float*)d_in[6];
  const float* dt_bias    = (const float*)d_in[7];
  const float* A_log      = (const float*)d_in[8];
  const float* D_param    = (const float*)d_in[9];
  const float* gated_w    = (const float*)d_in[10];
  const float* out_base   = (const float*)d_in[11];
  const float* lora_A_out = (const float*)d_in[12];
  const float* lora_B_out = (const float*)d_in[13];
  const float* loop_w     = (const float*)d_in[14];
  const float* lifeline   = (const float*)d_in[15];
  const int*   n_loops    = (const int*)d_in[16];
  float* out = (float*)d_out;

  // workspace layout (all sizes multiples of 256 bytes)
  char* ws = (char*)d_ws;
  size_t off = 0;
  f16_t* Win_f16  = (f16_t*)(ws + off); off += (size_t)D_IN_PAD * D_MODEL * 2;  // 5.0 MB
  f16_t* Wout_f16 = (f16_t*)(ws + off); off += (size_t)D_MODEL * D_INNER * 2;   // 2.36 MB
  f16_t* h_f16    = (f16_t*)(ws + off); off += (size_t)NTOK * D_MODEL * 2;      // 3.0 MB
  float* zx       = (float*)(ws + off); off += (size_t)NTOK * D_IN_PAD * 4;     // 26.7 MB
  float* xBC      = (float*)(ws + off); off += (size_t)NTOK * CONV_DIM * 4;     // 13.6 MB
  float* dtb      = (float*)(ws + off); off += (size_t)NTOK * NHEADS * 4;
  float* dAb      = (float*)(ws + off); off += (size_t)NTOK * NHEADS * 4;
  float* ybuf     = (float*)(ws + off); off += (size_t)NTOK * D_INNER * 4;      // 12.6 MB
  f16_t* yg_f16   = (f16_t*)(ws + off); off += (size_t)NTOK * D_INNER * 2;      // 6.3 MB
  float* g2out    = (float*)(ws + off); off += (size_t)NTOK * D_MODEL * 4;      // 6.3 MB
  float* xcur     = (float*)(ws + off); off += (size_t)NTOK * D_MODEL * 4;      // 6.3 MB
  (void)off; (void)ws_size; (void)in_sizes; (void)n_in; (void)out_size;

  // LoRA-merged weights (loop-invariant)
  prep_win_kernel<<<(D_IN_PAD * D_MODEL + 255) / 256, 256, 0, stream>>>(
      in_base, lora_A_in, lora_B_in, Win_f16);
  prep_wout_kernel<<<(D_MODEL * D_INNER + 255) / 256, 256, 0, stream>>>(
      out_base, lora_A_out, lora_B_out, Wout_f16);

  for (int loop = 0; loop < MAX_LOOPS; loop++) {
    const float* xin = (loop == 0) ? x : xcur;

    rmsnorm_rope_kernel<<<NTOK, 384, 0, stream>>>(
        xin, loop_w, h_f16, loop, n_loops);

    // zxbcdt = h @ W_in^T : M=2048, N=3264(pad), K=768
    {
      dim3 grid(D_IN_PAD / 64, NTOK / 32);
      gemm_wmma_f16_kernel<<<grid, 32, 0, stream>>>(
          h_f16, Win_f16, zx, D_IN_PAD, D_MODEL, loop, n_loops);
    }

    dt_da_kernel<<<(NTOK * NHEADS + 255) / 256, 256, 0, stream>>>(
        zx, dt_bias, A_log, dtb, dAb, loop, n_loops);

    conv_silu_kernel<<<(NTOK * CONV_DIM + 255) / 256, 256, 0, stream>>>(
        zx, conv_w, conv_b, xBC, loop, n_loops);

    ssm_scan_kernel<<<BATCH * NHEADS, 64, 0, stream>>>(
        xBC, dtb, dAb, D_param, ybuf, loop, n_loops);

    gate_norm_kernel<<<NTOK, 256, 0, stream>>>(
        ybuf, zx, gated_w, yg_f16, loop, n_loops);

    // out = yg @ W_out^T : M=2048, N=768, K=1536
    {
      dim3 grid(D_MODEL / 64, NTOK / 32);
      gemm_wmma_f16_kernel<<<grid, 32, 0, stream>>>(
          yg_f16, Wout_f16, g2out, D_MODEL, D_INNER, loop, n_loops);
    }

    residual_kernel<<<(NTOK * D_MODEL + 255) / 256, 256, 0, stream>>>(
        g2out, x_prompt, lifeline, xcur, loop, n_loops);
  }

  finalize_kernel<<<(NTOK * D_MODEL + 255) / 256, 256, 0, stream>>>(
      xcur, x, out, n_loops);
}
